// SLMuSELossModule_17763984736998
// MI455X (gfx1250) — matrix operand order repaired and checked by reading, not compiled, not run on hardware
//
#include <hip/hip_runtime.h>
#include <hip/hip_bf16.h>
#include <math.h>

// ---------------------------------------------------------------------------
// SLMuSE loss for MI455X (gfx1250, wave32, WMMA).
// Heavy term: vhat @ negatives^T (4096x1024x768) via V_WMMA_F32_16X16X32_BF16,
// bf16x3 fp32-emulation (hi*hi + lo*hi + hi*lo), f32 accumulate.
// LDS-staged, double-buffered: block converts f32->bf16(hi,lo) once into LDS,
// waves feed fragments with ds_load_b128. 32x32 per-wave tile = 12 WMMA/kstep.
// ---------------------------------------------------------------------------

#define D_DIM 768
#define EPSF  1e-8f

typedef __attribute__((ext_vector_type(16))) __bf16 v16bf;
typedef __attribute__((ext_vector_type(8)))  __bf16 v8bf;
typedef __attribute__((ext_vector_type(4)))  __bf16 v4bf;
typedef __attribute__((ext_vector_type(8)))  float  v8f;

struct bf16pair { __bf16 hi; __bf16 lo; };

// fp32 -> bf16 (RNE) split via hardware converters: x ~= hi + lo.
__device__ inline bf16pair split_bf16(float x) {
  bf16pair p;
  p.hi = (__bf16)x;                 // v_cvt (RNE)
  p.lo = (__bf16)(x - (float)p.hi); // residual, also RNE
  return p;
}

union V16u { v16bf v; v8bf h[2]; };

// A-fragment from LDS: lane l -> row, half h=l>>4; element i covers
// K = 16*h + i  -> one contiguous 16-element (32B) run.
__device__ inline v16bf ld_fragA(const __bf16* base, int row, int h) {
  return *(const v16bf*)(base + row * 32 + 16 * h);
}

// B-fragment from LDS: lane l -> col, half h; element i covers K = 16*h + i
// -> one contiguous 16-element (32B) run.
__device__ inline v16bf ld_fragB(const __bf16* base, int col, int h) {
  return *(const v16bf*)(base + col * 32 + 16 * h);
}

__device__ inline v8f wmma_bf16x3(v8f acc, const v16bf& ahi, const v16bf& alo,
                                  const v16bf& bhi, const v16bf& blo) {
  acc = __builtin_amdgcn_wmma_f32_16x16x32_bf16(false, ahi, false, bhi, (short)0, acc, false, false);
  acc = __builtin_amdgcn_wmma_f32_16x16x32_bf16(false, alo, false, bhi, (short)0, acc, false, false);
  acc = __builtin_amdgcn_wmma_f32_16x16x32_bf16(false, ahi, false, blo, (short)0, acc, false, false);
  return acc;
}

__device__ inline void split_store4(const float4 f, __bf16* hi, __bf16* lo) {
  v4bf hv, lv;
  bf16pair p0 = split_bf16(f.x); hv[0] = p0.hi; lv[0] = p0.lo;
  bf16pair p1 = split_bf16(f.y); hv[1] = p1.hi; lv[1] = p1.lo;
  bf16pair p2 = split_bf16(f.z); hv[2] = p2.hi; lv[2] = p2.lo;
  bf16pair p3 = split_bf16(f.w); hv[3] = p3.hi; lv[3] = p3.lo;
  *(v4bf*)hi = hv;
  *(v4bf*)lo = lv;
}

// ws layout (floats): [0]=Ju_sum [1]=Jt_sum [2]=ortho_sum [3]=mask_sum
//                     [4, 4+B)          true_d
//                     [4+B, 4+2B)       |vhat|^2
//                     [4+2B, 4+2B+N)    |neg|^2
__global__ void init_ws_kernel(float* ws) {
  if (threadIdx.x == 0) { ws[0] = 0.f; ws[1] = 0.f; ws[2] = 0.f; ws[3] = 0.f; }
}

__global__ __launch_bounds__(256) void row_stats_kernel(
    const float* __restrict__ v, const float* __restrict__ vhat,
    const float* __restrict__ negs, const int* __restrict__ mask,
    float* ws, int B, int N) {
  const int w = threadIdx.x >> 5, lane = threadIdx.x & 31;
  const int gw = blockIdx.x * 8 + w;
  float* true_d = ws + 4;
  float* vn2    = ws + 4 + B;
  float* nn2    = ws + 4 + 2 * B;
  if (gw < B) {
    const float* pw = vhat + (size_t)gw * D_DIM;
    const float* pv = v    + (size_t)gw * D_DIM;
    float n2 = 0.f, dd = 0.f;
    for (int d = lane; d < D_DIM; d += 32) {
      float a = pw[d], b = pv[d];
      n2 = fmaf(a, a, n2);
      float df = a - b;
      dd = fmaf(df, df, dd);
    }
    for (int off = 16; off; off >>= 1) {
      n2 += __shfl_xor(n2, off, 32);
      dd += __shfl_xor(dd, off, 32);
    }
    if (lane == 0) {
      vn2[gw]    = n2;
      true_d[gw] = sqrtf(fmaxf(dd, 0.f) + EPSF);
      atomicAdd(ws + 3, (float)mask[gw]);
    }
  } else if (gw < B + N) {
    const int n = gw - B;
    const float* p = negs + (size_t)n * D_DIM;
    float n2 = 0.f;
    for (int d = lane; d < D_DIM; d += 32) { float a = p[d]; n2 = fmaf(a, a, n2); }
    for (int off = 16; off; off >>= 1) n2 += __shfl_xor(n2, off, 32);
    if (lane == 0) nn2[n] = n2;
  }
}

// Contrastive term: block tile = 64 rows x 128 cols, K-step 32, LDS staged.
// 8 waves (2 row-groups x 4 col-groups), each wave: 32x32 tile, 2x2 accums.
__global__ __launch_bounds__(256) void ju_kernel(
    const float* __restrict__ vhat, const float* __restrict__ negs,
    const int* __restrict__ mask, float* ws, int B, int N) {
  const int tid  = threadIdx.x;
  const int w    = tid >> 5, lane = tid & 31;
  const int h    = lane >> 4, lid = lane & 15;
  const int rowBlock = blockIdx.y * 64;
  const int colBlock = blockIdx.x * 128;

  __shared__ __align__(32) __bf16 sAhi[2][64 * 32];
  __shared__ __align__(32) __bf16 sAlo[2][64 * 32];
  __shared__ __align__(32) __bf16 sBhi[2][128 * 32];
  __shared__ __align__(32) __bf16 sBlo[2][128 * 32];

  const int ldr = tid >> 3;       // 0..31 : row within pass
  const int kq  = (tid & 7) * 4;  // float4 column offset within 32-wide K slab

  // local fragment coordinates
  const int aRow0 = (w >> 2) * 32 + lid;
  const int aRow1 = aRow0 + 16;
  const int bCol0 = (w & 3) * 32 + lid;
  const int bCol1 = bCol0 + 16;

  const int NK = D_DIM / 32;      // 24 k-steps

  // ---- stage k-step 0 into buffer 0 ----
#pragma unroll
  for (int p = 0; p < 2; ++p) {
    const int r = p * 32 + ldr;
    float4 f = *(const float4*)(vhat + (size_t)(rowBlock + r) * D_DIM + kq);
    split_store4(f, &sAhi[0][r * 32 + kq], &sAlo[0][r * 32 + kq]);
  }
#pragma unroll
  for (int p = 0; p < 4; ++p) {
    const int r = p * 32 + ldr;
    float4 f = *(const float4*)(negs + (size_t)(colBlock + r) * D_DIM + kq);
    split_store4(f, &sBhi[0][r * 32 + kq], &sBlo[0][r * 32 + kq]);
  }
  __syncthreads();

  v8f acc00 = {}, acc01 = {}, acc10 = {}, acc11 = {};
  for (int ks = 0; ks < NK; ++ks) {
    const int cur = ks & 1;
    // ---- stage next k-step into the other buffer (overlaps with WMMA) ----
    if (ks + 1 < NK) {
      const int nxt = cur ^ 1;
      const int k0 = (ks + 1) * 32;
#pragma unroll
      for (int p = 0; p < 2; ++p) {
        const int r = p * 32 + ldr;
        float4 f = *(const float4*)(vhat + (size_t)(rowBlock + r) * D_DIM + k0 + kq);
        split_store4(f, &sAhi[nxt][r * 32 + kq], &sAlo[nxt][r * 32 + kq]);
      }
#pragma unroll
      for (int p = 0; p < 4; ++p) {
        const int r = p * 32 + ldr;
        float4 f = *(const float4*)(negs + (size_t)(colBlock + r) * D_DIM + k0 + kq);
        split_store4(f, &sBhi[nxt][r * 32 + kq], &sBlo[nxt][r * 32 + kq]);
      }
    }
    // ---- consume current buffer: 12 WMMAs ----
    v16bf a0h = ld_fragA(sAhi[cur], aRow0, h);
    v16bf a0l = ld_fragA(sAlo[cur], aRow0, h);
    v16bf a1h = ld_fragA(sAhi[cur], aRow1, h);
    v16bf a1l = ld_fragA(sAlo[cur], aRow1, h);
    v16bf b0h = ld_fragB(sBhi[cur], bCol0, h);
    v16bf b0l = ld_fragB(sBlo[cur], bCol0, h);
    v16bf b1h = ld_fragB(sBhi[cur], bCol1, h);
    v16bf b1l = ld_fragB(sBlo[cur], bCol1, h);
    acc00 = wmma_bf16x3(acc00, a0h, a0l, b0h, b0l);
    acc01 = wmma_bf16x3(acc01, a0h, a0l, b1h, b1l);
    acc10 = wmma_bf16x3(acc10, a1h, a1l, b0h, b0l);
    acc11 = wmma_bf16x3(acc11, a1h, a1l, b1h, b1l);
    __syncthreads();
  }

  // ---- epilogue: d2 -> sqrt -> hinge -> masked sum ----
  const float* true_d = ws + 4;
  const float* vn2    = ws + 4 + B;
  const float* nn2    = ws + 4 + 2 * B;
  const int rowBase = rowBlock + (w >> 2) * 32;
  const int colBase = colBlock + (w & 3) * 32;
  float s = 0.f;
#pragma unroll
  for (int ct = 0; ct < 2; ++ct) {
    const float nn = nn2[colBase + 16 * ct + lid];
#pragma unroll
    for (int rt = 0; rt < 2; ++rt) {
      const v8f* ap = (rt == 0) ? ((ct == 0) ? &acc00 : &acc01)
                                : ((ct == 0) ? &acc10 : &acc11);
#pragma unroll
      for (int r = 0; r < 8; ++r) {
        const int m = rowBase + 16 * rt + r + 8 * h;   // C/D: VGPR r, half h -> row
        float d2 = vn2[m] + nn - 2.f * (*ap)[r];
        float nd = sqrtf(fmaxf(d2, 0.f) + EPSF);
        float c  = fmaxf(1.f + true_d[m] - nd, 0.f);
        s += c * (float)mask[m];
      }
    }
  }
  for (int off = 16; off; off >>= 1) s += __shfl_xor(s, off, 32);
  __shared__ float red[8];
  if (lane == 0) red[w] = s;
  __syncthreads();
  if (tid == 0) {
    float t = 0.f;
#pragma unroll
    for (int i = 0; i < 8; ++i) t += red[i];
    atomicAdd(ws + 0, t);
  }
}

// Focal triplet: one wave per row; top-8 smallest g via shuffle argmin (wave32).
__global__ __launch_bounds__(256) void jt_kernel(
    const float* __restrict__ vhat, const float* __restrict__ g,
    const float* __restrict__ F, const int* __restrict__ mask,
    float* ws, int B) {
  const int w = threadIdx.x >> 5, lane = threadIdx.x & 31;
  const int b = blockIdx.x * 8 + w;
  if (b >= B) return;

  float v0 = g[(size_t)b * 64 + lane];
  float v1 = g[(size_t)b * 64 + 32 + lane];
  const int i0 = lane, i1 = 32 + lane;
  float gt[8]; int gi[8];
#pragma unroll
  for (int t = 0; t < 8; ++t) {
    float cv; int ci;
    if (v0 < v1 || (v0 == v1 && i0 < i1)) { cv = v0; ci = i0; } else { cv = v1; ci = i1; }
    for (int off = 16; off; off >>= 1) {
      float ov = __shfl_xor(cv, off, 32);
      int   oi = __shfl_xor(ci, off, 32);
      if (ov < cv || (ov == cv && oi < ci)) { cv = ov; ci = oi; }
    }
    gt[t] = cv; gi[t] = ci;
    if (ci == i0) v0 = 3.4e38f;
    if (ci == i1) v1 = 3.4e38f;
  }

  float gs = 1e-10f;
#pragma unroll
  for (int t = 0; t < 8; ++t) gs += gt[t];
  const float td = (ws + 4)[b];
  const float* pv = vhat + (size_t)b * D_DIM;
  float jrow = 0.f;
#pragma unroll
  for (int t = 0; t < 8; ++t) {
    const float* pf = F + (size_t)gi[t] * D_DIM;
    float dd = 0.f;
    for (int d = lane; d < D_DIM; d += 32) {
      float df = pv[d] - pf[d];
      dd = fmaf(df, df, dd);
    }
    for (int off = 16; off; off >>= 1) dd += __shfl_xor(dd, off, 32);
    float nd = sqrtf(fmaxf(dd, 0.f) + EPSF);
    float gn = gt[t] / gs;
    float mt = (1.f - gn) * (1.f - gn);          // MARGIN = 1.0
    jrow += fmaxf(mt + td - nd, 0.f);
  }
  jrow *= 0.125f * (float)mask[b];
  if (lane == 0) atomicAdd(ws + 1, jrow);
}

// ||F F^T - I||_F^2 : 64x64 gram = 16 WMMA tiles, single block of 8 waves.
__global__ __launch_bounds__(256) void ortho_kernel(const float* __restrict__ F, float* ws) {
  const int w = threadIdx.x >> 5, lane = threadIdx.x & 31;
  const int h = lane >> 4, lid = lane & 15;
  float s = 0.f;
  for (int it = 0; it < 2; ++it) {
    const int t = w + it * 8;
    const int rowBase = (t >> 2) * 16, colBase = (t & 3) * 16;
    v8f acc = {};
    for (int k0 = 0; k0 < D_DIM; k0 += 32) {
      v16bf ahi, alo, bhi, blo;
      // A fragment direct from global (tiny problem): run 16h
      {
        const float* rowp = F + (size_t)(rowBase + lid) * D_DIM + k0;
#pragma unroll
        for (int i = 0; i < 8; ++i) {
          bf16pair p = split_bf16(rowp[16 * h + i]);
          ahi[i] = p.hi; alo[i] = p.lo;
        }
#pragma unroll
        for (int i = 0; i < 8; ++i) {
          bf16pair p = split_bf16(rowp[16 * h + 8 + i]);
          ahi[8 + i] = p.hi; alo[8 + i] = p.lo;
        }
      }
      {
        const float* colp = F + (size_t)(colBase + lid) * D_DIM + k0 + 16 * h;
#pragma unroll
        for (int i = 0; i < 16; ++i) {
          bf16pair p = split_bf16(colp[i]);
          bhi[i] = p.hi; blo[i] = p.lo;
        }
      }
      acc = wmma_bf16x3(acc, ahi, alo, bhi, blo);
    }
    const int gn = colBase + lid;
#pragma unroll
    for (int r = 0; r < 8; ++r) {
      const int gm = rowBase + r + 8 * h;
      float dd = acc[r] - (gm == gn ? 1.f : 0.f);
      s = fmaf(dd, dd, s);
    }
  }
  for (int off = 16; off; off >>= 1) s += __shfl_xor(s, off, 32);
  __shared__ float red[8];
  if (lane == 0) red[w] = s;
  __syncthreads();
  if (threadIdx.x == 0) {
    float t = 0.f;
#pragma unroll
    for (int i = 0; i < 8; ++i) t += red[i];
    atomicAdd(ws + 2, t);
  }
}

__global__ void finalize_kernel(const float* __restrict__ ws, float* __restrict__ out, int N) {
  if (threadIdx.x == 0 && blockIdx.x == 0) {
    float ms = ws[3];
    if (ms < 1.f) ms = 1.f;                       // clip(mask.sum(), 1.0)
    out[0] = ws[0] / (ms * (float)N) + ws[1] / ms + 0.01f * ws[2];
  }
}

extern "C" void kernel_launch(void* const* d_in, const int* in_sizes, int n_in,
                              void* d_out, int out_size, void* d_ws, size_t ws_size,
                              hipStream_t stream) {
  const float* v    = (const float*)d_in[0];
  const float* vhat = (const float*)d_in[1];
  const float* g    = (const float*)d_in[2];
  const float* F    = (const float*)d_in[3];
  const float* negs = (const float*)d_in[4];
  const int*   mask = (const int*)d_in[5];
  float* out = (float*)d_out;
  float* ws  = (float*)d_ws;

  const int B = in_sizes[0] / D_DIM;   // 4096
  const int N = in_sizes[4] / D_DIM;   // 1024

  init_ws_kernel<<<1, 32, 0, stream>>>(ws);

  const int rowsTotal = B + N;
  row_stats_kernel<<<(rowsTotal + 7) / 8, 256, 0, stream>>>(v, vhat, negs, mask, ws, B, N);

  dim3 juGrid(N / 128, B / 64);
  ju_kernel<<<juGrid, 256, 0, stream>>>(vhat, negs, mask, ws, B, N);

  jt_kernel<<<(B + 7) / 8, 256, 0, stream>>>(vhat, g, F, mask, ws, B);

  ortho_kernel<<<1, 256, 0, stream>>>(F, ws);

  finalize_kernel<<<1, 32, 0, stream>>>(ws, out, N);
}
